// GCN_17540646436881
// MI455X (gfx1250) — compile-verified, matrix operand order checked by reading, not run on hardware
//
#include <hip/hip_runtime.h>

#define N_NODES 100000
#define N_EDGES 1600000
#define C 128
#define M_TILES (N_NODES / 16)   // 6250, exact

typedef _Float16 v16h __attribute__((ext_vector_type(16)));
typedef _Float16 v8h  __attribute__((ext_vector_type(8)));
typedef float    v8f  __attribute__((ext_vector_type(8)));

union V16H {
    v16h v;
    v8h  half8[2];
    _Float16 h[16];
};

// ---------------------------------------------------------------------------
// Degree / normalization helpers
// ---------------------------------------------------------------------------
__global__ __launch_bounds__(256) void zero_f32(float* __restrict__ p, int n) {
    int i = blockIdx.x * blockDim.x + threadIdx.x;
    if (i < n) p[i] = 0.0f;
}

__global__ __launch_bounds__(256) void deg_accum(const int* __restrict__ ei,
                                                 const float* __restrict__ ew,
                                                 float* __restrict__ deg) {
    int e = blockIdx.x * blockDim.x + threadIdx.x;
    if (e < N_EDGES) atomicAdd(&deg[ei[N_EDGES + e]], ew[e]);
}

// deg -> deg^{-1/2}, with the self-loop weight (+1) folded in (deg+1 > 0 always)
__global__ __launch_bounds__(256) void deg_to_dinv(float* __restrict__ d) {
    int i = blockIdx.x * blockDim.x + threadIdx.x;
    if (i < N_NODES) d[i] = __frsqrt_rn(d[i] + 1.0f);
}

// ---------------------------------------------------------------------------
// GEMM: Y[N x 128] = relu?(X)[N x 128] @ W[128 x 128] via v_wmma_f32_16x16x32_f16
// One workgroup = 8 waves; wave w owns rows [(blk*8+w)*16, +16) and sweeps all
// 8 column tiles, reusing its A fragment across 8 WMMAs per 32-wide K chunk.
// W is staged transposed (WT[n][k]) in LDS as f16 so B fragments are two
// contiguous ds_load_b128 per (n, kchunk). RELU is a compile-time flag so the
// A conversion path is pure v_max/v_cvt_pk_f16_f32 with no per-element selects.
// ---------------------------------------------------------------------------
template <int RELU>
__global__ __launch_bounds__(256) void gcn_gemm_wmma(const float* __restrict__ X,
                                                     const float* __restrict__ W,
                                                     float* __restrict__ Y) {
    __shared__ _Float16 WT[C * C];  // 32 KB: WT[n*128 + k] = (f16)W[k*128 + n]

    const int t = threadIdx.x;
    // Stage + transpose + downconvert W: coalesced global reads, scattered LDS writes.
    #pragma unroll
    for (int i = 0; i < (C * C) / 256; ++i) {
        int j = t + i * 256;            // flat index into row-major W
        float w = W[j];
        int k = j >> 7;
        int n = j & (C - 1);
        WT[n * C + k] = (_Float16)w;
    }
    __syncthreads();

    const int wave  = t >> 5;
    const int lane  = t & 31;
    const int mTile = blockIdx.x * 8 + wave;
    if (mTile >= M_TILES) return;       // wave-uniform; EXEC stays all-ones for WMMA

    const int laneLo = lane & 15;
    const int hi     = lane >> 4;       // ISA 16-bit A/B layout: hi-half lanes take K+8
    const int row    = mTile * 16 + laneLo;
    const float* xrow = X + (size_t)row * C;

    v8f acc[8];
    #pragma unroll
    for (int n = 0; n < 8; ++n) acc[n] = (v8f){0, 0, 0, 0, 0, 0, 0, 0};

    #pragma unroll
    for (int kk = 0; kk < C; kk += 32) {
        const int k0 = kk + hi * 8;     // elements 0..7  -> K = k0..k0+7
        const int k1 = k0 + 16;         // elements 8..15 -> K = k1..k1+7

        // ---- A fragment (16x32 f16): per-lane 4x float4 loads + cvt (+ fused ReLU)
        float4 f0 = *(const float4*)(xrow + k0);
        float4 f1 = *(const float4*)(xrow + k0 + 4);
        float4 f2 = *(const float4*)(xrow + k1);
        float4 f3 = *(const float4*)(xrow + k1 + 4);
        float af[16] = {f0.x, f0.y, f0.z, f0.w, f1.x, f1.y, f1.z, f1.w,
                        f2.x, f2.y, f2.z, f2.w, f3.x, f3.y, f3.z, f3.w};
        V16H a;
        #pragma unroll
        for (int e = 0; e < 16; ++e) {
            float v = af[e];
            if (RELU) v = fmaxf(v, 0.0f);   // compile-time: no cndmask chains
            a.h[e] = (_Float16)v;
        }

        // ---- Sweep all 8 N tiles reusing A
        #pragma unroll
        for (int n = 0; n < 8; ++n) {
            const _Float16* wt = WT + (size_t)(n * 16 + laneLo) * C;
            V16H b;
            b.half8[0] = *(const v8h*)(wt + k0);   // ds_load_b128
            b.half8[1] = *(const v8h*)(wt + k1);   // ds_load_b128
            acc[n] = __builtin_amdgcn_wmma_f32_16x16x32_f16(
                false, a.v, false, b.v, (short)0, acc[n], false, false);
        }
    }

    // ---- Store: C/D layout -> VGPR r holds (M = r + hi*8, N = laneLo) of each tile
    float* ybase = Y + (size_t)(mTile * 16) * C;
    #pragma unroll
    for (int n = 0; n < 8; ++n) {
        #pragma unroll
        for (int r = 0; r < 8; ++r) {
            ybase[(size_t)(r + hi * 8) * C + n * 16 + laneLo] = acc[n][r];
        }
    }
}

// ---------------------------------------------------------------------------
// Aggregation init: out[i][:] = b + xw[i][:] * dinv[i]^2   (self-loop + bias)
// Also serves as the deterministic re-zero of the accumulation buffer.
// ---------------------------------------------------------------------------
__global__ __launch_bounds__(256) void agg_init(const float* __restrict__ xw,
                                                const float* __restrict__ dinv,
                                                const float* __restrict__ b,
                                                float* __restrict__ out) {
    int i    = blockIdx.x * blockDim.x + threadIdx.x;   // N_NODES * 32 threads
    int node = i >> 5;
    int c4   = (i & 31) * 4;
    if (node >= N_NODES) return;
    float di = dinv[node];
    float s  = di * di;
    float4 v  = *(const float4*)(xw + (size_t)node * C + c4);
    float4 bb = *(const float4*)(b + c4);
    float4 o  = {bb.x + v.x * s, bb.y + v.y * s, bb.z + v.z * s, bb.w + v.w * s};
    *(float4*)(out + (size_t)node * C + c4) = o;
}

// ---------------------------------------------------------------------------
// Edge scatter: one wave per edge; lane gathers float4 of xw[row], scales by
// norm = dinv[row]*ew*dinv[col], scatters with global_atomic_add_f32.
// xw (51.2 MB) and out (51.2 MB) both fit the 192 MB L2, so gather + atomics
// stay on-die; only edge lists stream from HBM.
// ---------------------------------------------------------------------------
__global__ __launch_bounds__(256) void agg_edges(const int* __restrict__ ei,
                                                 const float* __restrict__ ew,
                                                 const float* __restrict__ xw,
                                                 const float* __restrict__ dinv,
                                                 float* __restrict__ out) {
    int tid  = blockIdx.x * blockDim.x + threadIdx.x;
    int e    = tid >> 5;
    int lane = tid & 31;
    if (e >= N_EDGES) return;
    int r = ei[e];
    int c = ei[N_EDGES + e];
    float norm = dinv[r] * ew[e] * dinv[c];
    const float* src = xw + (size_t)r * C + lane * 4;
    float*       dst = out + (size_t)c * C + lane * 4;
    float4 v = *(const float4*)src;
    atomicAdd(dst + 0, v.x * norm);
    atomicAdd(dst + 1, v.y * norm);
    atomicAdd(dst + 2, v.z * norm);
    atomicAdd(dst + 3, v.w * norm);
}

// ---------------------------------------------------------------------------
extern "C" void kernel_launch(void* const* d_in, const int* in_sizes, int n_in,
                              void* d_out, int out_size, void* d_ws, size_t ws_size,
                              hipStream_t stream) {
    const float* x  = (const float*)d_in[0];
    const int*   ei = (const int*)  d_in[1];   // [2, E] row-major: rows then cols
    const float* ew = (const float*)d_in[2];
    const float* W1 = (const float*)d_in[3];
    const float* b1 = (const float*)d_in[4];
    const float* W2 = (const float*)d_in[5];
    const float* b2 = (const float*)d_in[6];
    const float* W3 = (const float*)d_in[7];
    const float* b3 = (const float*)d_in[8];
    float* out = (float*)d_out;

    float* dinv = (float*)d_ws;                       // N floats
    float* xw   = dinv + N_NODES;                     // N*C floats
    float* hbuf = xw + (size_t)N_NODES * C;           // N*C floats

    const int B = 256;
    dim3 blk(B);

    // deg^{-1/2} (shared by all three layers)
    zero_f32<<<dim3((N_NODES + B - 1) / B), blk, 0, stream>>>(dinv, N_NODES);
    deg_accum<<<dim3((N_EDGES + B - 1) / B), blk, 0, stream>>>(ei, ew, dinv);
    deg_to_dinv<<<dim3((N_NODES + B - 1) / B), blk, 0, stream>>>(dinv);

    const dim3 gemmGrid((M_TILES + 7) / 8);               // 782 workgroups
    const dim3 initGrid((N_NODES * 32 + B - 1) / B);      // 12500
    const dim3 edgeGrid(((size_t)N_EDGES * 32) / B);      // 200000

    // Layer 1
    gcn_gemm_wmma<0><<<gemmGrid, blk, 0, stream>>>(x, W1, xw);
    agg_init<<<initGrid, blk, 0, stream>>>(xw, dinv, b1, hbuf);
    agg_edges<<<edgeGrid, blk, 0, stream>>>(ei, ew, xw, dinv, hbuf);

    // Layer 2 (ReLU of layer-1 output fused into the GEMM A-read)
    gcn_gemm_wmma<1><<<gemmGrid, blk, 0, stream>>>(hbuf, W2, xw);
    agg_init<<<initGrid, blk, 0, stream>>>(xw, dinv, b2, hbuf);
    agg_edges<<<edgeGrid, blk, 0, stream>>>(ei, ew, xw, dinv, hbuf);

    // Layer 3 (ReLU of layer-2 output fused; no ReLU after — aggregate into d_out)
    gcn_gemm_wmma<1><<<gemmGrid, blk, 0, stream>>>(hbuf, W3, xw);
    agg_init<<<initGrid, blk, 0, stream>>>(xw, dinv, b3, out);
    agg_edges<<<edgeGrid, blk, 0, stream>>>(ei, ew, xw, dinv, out);
}
